// QLoRABigNet_65240553226720
// MI455X (gfx1250) — matrix operand order repaired and checked by reading, not compiled, not run on hardware
//
#include <hip/hip_runtime.h>
#include <hip/hip_bf16.h>

// CDNA5 / gfx1250 QLoRA BigNet forward.
// W_eff = dequant(qw)*scale + lb@la folded once, stored in a WMMA-B-fragment
// swizzled f16 layout; 18 WMMA GEMMs (fused bias/ReLU/residual) + 5 LayerNorms.
// GEMM staging uses GLOBAL_LOAD_ASYNC_TO_LDS_B128 (ASYNCcnt) with a 3-deep
// LDS pipeline when the builtin is available; sync staging otherwise.

typedef __attribute__((ext_vector_type(16))) _Float16 v16h;
typedef __attribute__((ext_vector_type(8)))  float    v8f;
typedef int vint4_t __attribute__((vector_size(16)));   // matches builtin param

#define DIM 1024
#define NROWS 8192
#define NLAYERS 18

#ifdef __has_builtin
#if __has_builtin(__builtin_amdgcn_global_load_async_to_lds_b128)
#define HAVE_ASYNC_LDS 1
#endif
#endif
#ifndef HAVE_ASYNC_LDS
#define HAVE_ASYNC_LDS 0
#endif

#if HAVE_ASYNC_LDS
// INST_OFFSET is added to BOTH the global and LDS addresses (ISA 08 §4.4).
// Builtin params are (int4 as1*, int4 as3*, imm offset, imm cpol).
#define GLDS_B128(gptr, ldsptr, off)                                             \
    __builtin_amdgcn_global_load_async_to_lds_b128(                              \
        (__attribute__((address_space(1))) vint4_t*)(unsigned long long)(gptr),  \
        (__attribute__((address_space(3))) vint4_t*)(unsigned int)(unsigned long long)(ldsptr), \
        off, 0)
#endif

#ifdef __has_builtin
#if __has_builtin(__builtin_amdgcn_s_wait_asynccnt)
#define WAIT_ASYNC(n) __builtin_amdgcn_s_wait_asynccnt(n)
#endif
#endif
#ifndef WAIT_ASYNC
#define WAIT_ASYNC(n) asm volatile("s_wait_asynccnt %0" ::"i"(n) : "memory")
#endif

// ---------------------------------------------------------------------------
// Weight prep: fragment (l, kt, nf) = 512 contiguous f16.
// lane l5: n = nf*16 + (l5&15); 16 halves hold k = kt*32 + (l5>>4)*16 + h.
// ---------------------------------------------------------------------------
__global__ __launch_bounds__(256) void prep_weights_k(
    const int*   __restrict__ qw,      // [18,1024,1024] codes in [0,16)
    const float* __restrict__ scales,  // [18,1024,8]
    const float* __restrict__ la,      // [18,32,1024]
    const float* __restrict__ lb,      // [18,1024,32]
    _Float16*    __restrict__ wt)      // [18][32][64][512] f16
{
    const int t    = blockIdx.x * 256 + threadIdx.x;   // 18*32*64*32 threads
    const int lane = t & 31;
    const int nf   = (t >> 5)  & 63;
    const int kt   = (t >> 11) & 31;
    const int l    = t >> 16;
    const int n     = nf * 16 + (lane & 15);
    const int kbase = kt * 32 + (lane >> 4) * 16;

    const int*  qrow = qw + ((size_t)l * DIM + n) * DIM + kbase;
    const float sc   = scales[((size_t)l * DIM + n) * 8 + (kt >> 2)];

    float vals[16];
#pragma unroll
    for (int j = 0; j < 16; ++j)
        vals[j] = (float)(qrow[j] - 8) * sc;

    const float* lbrow  = lb + ((size_t)l * DIM + n) * 32;
    const float* labase = la + (size_t)l * 32 * DIM + kbase;
#pragma unroll 4
    for (int r = 0; r < 32; ++r) {
        const float bv = lbrow[r];
        const float* ar = labase + (size_t)r * DIM;
#pragma unroll
        for (int j = 0; j < 16; ++j)
            vals[j] += bv * ar[j];
    }

    alignas(16) _Float16 h[16];
#pragma unroll
    for (int j = 0; j < 16; ++j) h[j] = (_Float16)vals[j];

    _Float16* dst = wt + (size_t)t * 16;
    *(uint4*)(dst)     = *(const uint4*)(h);
    *(uint4*)(dst + 8) = *(const uint4*)(h + 8);
}

// ---------------------------------------------------------------------------
__global__ __launch_bounds__(256) void init_h_k(
    const float* __restrict__ x, float* __restrict__ H, _Float16* __restrict__ a0)
{
    const size_t i = ((size_t)blockIdx.x * 256 + threadIdx.x) * 4;
    const float4 v = *(const float4*)(x + i);
    *(float4*)(H + i) = v;
    alignas(8) _Float16 h[4] = {(_Float16)v.x, (_Float16)v.y, (_Float16)v.z, (_Float16)v.w};
    *(uint2*)(a0 + i) = *(const uint2*)h;
}

// ---------------------------------------------------------------------------
// Fused WMMA GEMM: out = [relu](act @ W_eff^T + bias) [+ resid]
// Tile 128x128, 256 threads = 8 wave32, wave tile 32x64 (2x4 16x16 frags).
// ---------------------------------------------------------------------------
__global__ __launch_bounds__(256) void gemm_qlora_k(
    const _Float16* __restrict__ act,   // [8192,1024] row-major f16
    const _Float16* __restrict__ wt,    // layer base, fragment-swizzled
    const float*    __restrict__ bias,  // [1024]
    const float*    __restrict__ resid, // [8192,1024] or null
    float*          __restrict__ outf,  // [8192,1024] or null
    _Float16*       __restrict__ outh,  // [8192,1024] or null
    int relu)
{
    __shared__ __align__(16) _Float16 As[3][128 * 40];  // pitch 40 halves (80B)
    __shared__ __align__(16) _Float16 Bs[3][8 * 512];   // 8 B-fragments, raw copy

    const int tid  = threadIdx.x;
    const int bn   = blockIdx.x;            // 0..7   (N tiles)
    const int bm   = blockIdx.y;            // 0..63  (M tiles)
    const int lane = tid & 31;
    const int w    = tid >> 5;
    const int wm   = (w & 3) * 32;          // wave M offset in tile
    const int wn   = (w >> 2) * 64;         // wave N offset in tile
    const int lm   = lane & 15;
    const int kh   = lane >> 4;

    v8f acc[2][4];
#pragma unroll
    for (int i = 0; i < 2; ++i)
#pragma unroll
        for (int j = 0; j < 4; ++j)
            acc[i][j] = (v8f){0.f, 0.f, 0.f, 0.f, 0.f, 0.f, 0.f, 0.f};

    const int r    = tid >> 1;              // A-tile row 0..127
    const int part = tid & 1;               // 16-half chunk of the row
    const _Float16* gA = act + ((size_t)(bm * 128 + r)) * DIM + part * 16;
    const _Float16* gB = wt + (size_t)bn * 8 * 512 + (size_t)tid * 16;

    auto compute = [&](int buf) {
        v16h afrag[2], bfrag[4];
#pragma unroll
        for (int fm = 0; fm < 2; ++fm) {
            const _Float16* p = &As[buf][(wm + fm * 16 + lm) * 40 + kh * 16];
            ((uint4*)&afrag[fm])[0] = *(const uint4*)(p);
            ((uint4*)&afrag[fm])[1] = *(const uint4*)(p + 8);
        }
#pragma unroll
        for (int fn = 0; fn < 4; ++fn) {
            const _Float16* p = &Bs[buf][((wn >> 4) + fn) * 512 + lane * 16];
            bfrag[fn] = *(const v16h*)p;
        }
#pragma unroll
        for (int fm = 0; fm < 2; ++fm)
#pragma unroll
            for (int fn = 0; fn < 4; ++fn)
                acc[fm][fn] = __builtin_amdgcn_wmma_f32_16x16x32_f16(
                    false, afrag[fm], false, bfrag[fn],
                    (short)0, acc[fm][fn], false, false);
    };

#if HAVE_ASYNC_LDS
    // --- async staging: 4 GLOBAL_LOAD_ASYNC_TO_LDS_B128 per thread/step ---
    auto stage = [&](int kt, int buf) {
        const _Float16* ga = gA + kt * 32;
        const _Float16* gb = gB + (size_t)kt * (64 * 512);
        const _Float16* pa = &As[buf][r * 40 + part * 16];
        const _Float16* pb = &Bs[buf][tid * 16];
        GLDS_B128(ga, pa, 0);
        GLDS_B128(ga, pa, 16);
        GLDS_B128(gb, pb, 0);
        GLDS_B128(gb, pb, 16);
    };

    stage(0, 0);
    stage(1, 1);
#pragma unroll 1
    for (int kt = 0; kt < 32; ++kt) {
        const int cur = kt % 3;
        // stage(kt) landed once <=4 remain (stage(kt+1) may still be in flight;
        // async ops complete in order). Last iteration: drain fully.
        if (kt < 31) { WAIT_ASYNC(4); } else { WAIT_ASYNC(0); }
        __syncthreads();                    // all waves' stage(kt) visible
        compute(cur);
        __syncthreads();                    // all waves done reading buf cur
        if (kt + 2 < 32) stage(kt + 2, (kt + 2) % 3);
    }
#else
    // --- fallback: sync staging, double buffered ---
    auto stage = [&](int kt, int buf) {
        const uint4 a0 = *(const uint4*)(gA + kt * 32);
        const uint4 a1 = *(const uint4*)(gA + kt * 32 + 8);
        const uint4 b0 = *(const uint4*)(gB + (size_t)kt * (64 * 512));
        const uint4 b1 = *(const uint4*)(gB + (size_t)kt * (64 * 512) + 8);
        _Float16* pa = &As[buf][r * 40 + part * 16];
        *(uint4*)(pa)     = a0;
        *(uint4*)(pa + 8) = a1;
        _Float16* pb = &Bs[buf][tid * 16];
        *(uint4*)(pb)     = b0;
        *(uint4*)(pb + 8) = b1;
    };

    stage(0, 0);
    __syncthreads();
#pragma unroll 1
    for (int kt = 0; kt < 32; ++kt) {
        const int cur = kt & 1;
        if (kt < 31) stage(kt + 1, cur ^ 1);
        compute(cur);
        __syncthreads();
    }
#endif

    // Epilogue. D layout: lane (lm,kh) holds col N=lm, rows M=v+8*kh.
    const int col_base = bn * 128 + wn + lm;
    const int row_base = bm * 128 + wm + kh * 8;
#pragma unroll
    for (int fm = 0; fm < 2; ++fm) {
#pragma unroll
        for (int fn = 0; fn < 4; ++fn) {
            const int col = col_base + fn * 16;
            const float bv = bias[col];
#pragma unroll
            for (int v = 0; v < 8; ++v) {
                const int row = row_base + fm * 16 + v;
                float val = acc[fm][fn][v] + bv;
                if (relu) val = fmaxf(val, 0.f);
                const size_t idx = (size_t)row * DIM + col;
                if (resid) val += resid[idx];
                if (outf) outf[idx] = val;
                if (outh) outh[idx] = (_Float16)val;
            }
        }
    }
}

// ---------------------------------------------------------------------------
__global__ __launch_bounds__(256) void layernorm_k(
    float* __restrict__ H, const float* __restrict__ g, const float* __restrict__ b,
    _Float16* __restrict__ outh)
{
    const int row = blockIdx.x, tid = threadIdx.x;
    const int c0 = tid * 4;
    const float4 v = *(const float4*)(H + (size_t)row * DIM + c0);
    float s  = v.x + v.y + v.z + v.w;
    float s2 = v.x * v.x + v.y * v.y + v.z * v.z + v.w * v.w;
#pragma unroll
    for (int off = 16; off > 0; off >>= 1) {
        s  += __shfl_xor(s,  off, 32);
        s2 += __shfl_xor(s2, off, 32);
    }
    __shared__ float sh1[8], sh2[8];
    if ((tid & 31) == 0) { sh1[tid >> 5] = s; sh2[tid >> 5] = s2; }
    __syncthreads();
    if (tid == 0) {
        float a = 0.f, c = 0.f;
#pragma unroll
        for (int i = 0; i < 8; ++i) { a += sh1[i]; c += sh2[i]; }
        sh1[0] = a; sh2[0] = c;
    }
    __syncthreads();
    const float mean = sh1[0] * (1.f / DIM);
    const float var  = sh2[0] * (1.f / DIM) - mean * mean;
    const float rinv = rsqrtf(var + 1e-5f);
    const float4 gg = *(const float4*)(g + c0);
    const float4 bb = *(const float4*)(b + c0);
    float4 o;
    o.x = (v.x - mean) * rinv * gg.x + bb.x;
    o.y = (v.y - mean) * rinv * gg.y + bb.y;
    o.z = (v.z - mean) * rinv * gg.z + bb.z;
    o.w = (v.w - mean) * rinv * gg.w + bb.w;
    *(float4*)(H + (size_t)row * DIM + c0) = o;
    alignas(8) _Float16 h[4] = {(_Float16)o.x, (_Float16)o.y, (_Float16)o.z, (_Float16)o.w};
    *(uint2*)(outh + (size_t)row * DIM + c0) = *(const uint2*)h;
}

// ---------------------------------------------------------------------------
extern "C" void kernel_launch(void* const* d_in, const int* in_sizes, int n_in,
                              void* d_out, int out_size, void* d_ws, size_t ws_size,
                              hipStream_t stream) {
    const float* x      = (const float*)d_in[0];
    const int*   qw     = (const int*)  d_in[1];
    const float* scales = (const float*)d_in[2];
    const float* bias   = (const float*)d_in[3];
    const float* la     = (const float*)d_in[4];
    const float* lb     = (const float*)d_in[5];
    const float* ln_g   = (const float*)d_in[6];
    const float* ln_b   = (const float*)d_in[7];
    float* out = (float*)d_out;
    char*  ws  = (char*)d_ws;

    const size_t WT_BYTES  = (size_t)NLAYERS * DIM * DIM * 2;    // 37.75 MB f16
    const size_t ACT_BYTES = (size_t)NROWS * DIM * 2;            // 16.78 MB f16
    _Float16* Wt  = (_Float16*)(ws);
    _Float16* A0v = (_Float16*)(ws + WT_BYTES);
    _Float16* A1v = (_Float16*)(ws + WT_BYTES + ACT_BYTES);
    float*    H   = (float*)   (ws + WT_BYTES + 2 * ACT_BYTES);  // 33.55 MB f32

    prep_weights_k<<<(NLAYERS * 32 * 64 * 32) / 256, 256, 0, stream>>>(
        qw, scales, la, lb, Wt);
    init_h_k<<<(NROWS * DIM) / (256 * 4), 256, 0, stream>>>(x, H, A0v);

    const size_t LSTR = (size_t)DIM * DIM;  // halves per layer
    dim3 grid(DIM / 128, NROWS / 128);      // (8, 64)
    for (int blk = 0; blk < 6; ++blk) {
        const int l0 = blk * 3;
        gemm_qlora_k<<<grid, 256, 0, stream>>>(
            A0v, Wt + (size_t)l0 * LSTR, bias + l0 * DIM,
            nullptr, nullptr, A1v, 1);
        gemm_qlora_k<<<grid, 256, 0, stream>>>(
            A1v, Wt + (size_t)(l0 + 1) * LSTR, bias + (l0 + 1) * DIM,
            nullptr, nullptr, A0v, 1);
        float* of = (blk == 5) ? out : H;
        gemm_qlora_k<<<grid, 256, 0, stream>>>(
            A0v, Wt + (size_t)(l0 + 2) * LSTR, bias + (l0 + 2) * DIM,
            H, of, nullptr, 0);
        if (blk < 5)
            layernorm_k<<<NROWS, 256, 0, stream>>>(
                H, ln_g + blk * DIM, ln_b + blk * DIM, A0v);
    }
}
